// PhysNet_66855460930134
// MI455X (gfx1250) — compile-verified, matrix operand order checked by reading, not run on hardware
//
#include <hip/hip_runtime.h>
#include <hip/hip_bf16.h>

typedef __bf16 bf16;
typedef __attribute__((ext_vector_type(16))) __bf16 v16bf;
typedef __attribute__((ext_vector_type(8)))  __bf16 v8bf;
typedef __attribute__((ext_vector_type(8)))  float  v8f;

#define F_DIM 128
#define K_DIM 64
#define NB 5
#define NRI 3
#define NRA 2

// NOTE: relies on N_ATOMS (10000) and N_PAIRS (320000) being multiples of 16,
// so every 16-row WMMA subtile with subtile base < nRows is fully in bounds.

// shifted softplus: softplus(x) - ln(2)
__device__ __forceinline__ float ssp(float v) {
    float sp = (v > 15.f) ? v : log1pf(__expf(v));
    return sp - 0.69314718055994531f;
}

// ---- WMMA fragment helpers (ISA 7.12.2 layouts, wave32) -------------------
// 16-bit A matrix 16x32 (MxK): lane l (0-15) holds M=l; lanes 16-31 hold the
// upper K-halves.  Per lane the 16 elements are two contiguous 8-wide K runs:
//   e=0..7  -> k = kBase + half*8 + e
//   e=8..15 -> k = kBase + 16 + half*8 + (e-8)

// Load A fragment from f32 row-major [nRows, F_DIM], optional activation.
template <int ACT>
__device__ __forceinline__ v16bf load_a_frag_f32(const float* __restrict__ A,
                                                 int rowBase, int kBase) {
    int l = threadIdx.x & 31;
    int half = l >> 4, mm = l & 15;
    const float* p = A + (size_t)(rowBase + mm) * F_DIM + kBase + half * 8;
    float tmp[16];
    *(float4*)&tmp[0]  = *(const float4*)(p);
    *(float4*)&tmp[4]  = *(const float4*)(p + 4);
    *(float4*)&tmp[8]  = *(const float4*)(p + 16);
    *(float4*)&tmp[12] = *(const float4*)(p + 20);
    v16bf a;
#pragma unroll
    for (int e = 0; e < 16; ++e) {
        float v = tmp[e];
        if (ACT) v = ssp(v);
        a[e] = (bf16)v;
    }
    return a;
}

// Load A fragment from bf16 row-major [nRows, K_DIM] (pair RBF matrix).
__device__ __forceinline__ v16bf load_a_frag_bf16(const bf16* __restrict__ A,
                                                  int rowBase, int kBase) {
    int l = threadIdx.x & 31;
    int half = l >> 4, mm = l & 15;
    const bf16* p = A + (size_t)(rowBase + mm) * K_DIM + kBase + half * 8;
    v8bf c0 = *(const v8bf*)(p);
    v8bf c1 = *(const v8bf*)(p + 16);
    return __builtin_shufflevector(c0, c1, 0, 1, 2, 3, 4, 5, 6, 7,
                                   8, 9, 10, 11, 12, 13, 14, 15);
}

// Fill LDS with the weight matrix pre-swizzled into B-fragment layout:
// Wt[((ks*8 + nt)*32 + lane)*16 + e]  ==  W[(ks*32 + frag_k(e,lane>>4)) * F_DIM
//                                          + nt*16 + (lane&15)]
// Each 8-element chunk (fixed ks/h2/half/ncol) is 8 consecutive k rows at one
// column -> coalesced-across-threads global reads, one ds_store_b128 each.
template <int KSTEPS>
__device__ __forceinline__ void fill_wfrag(const float* __restrict__ W,
                                           bf16* __restrict__ Wt, int tid) {
#pragma unroll 2
    for (int c = tid; c < KSTEPS * 512; c += 128) {
        int ncol = c & 127;
        int r = c >> 7;
        int half = r & 1, h2 = (r >> 1) & 1, ks = r >> 2;
        int kbase = ks * 32 + h2 * 16 + half * 8;
        v8bf vb;
#pragma unroll
        for (int j = 0; j < 8; ++j) vb[j] = (bf16)W[(size_t)(kbase + j) * F_DIM + ncol];
        int lane = half * 16 + (ncol & 15);
        int nt = ncol >> 4;
        *(v8bf*)&Wt[(size_t)(((ks * 8 + nt) * 32 + lane) << 4) + h2 * 8] = vb;
    }
}

__device__ __forceinline__ v16bf load_b_frag(const bf16* __restrict__ Wt, int ks, int nt) {
    int lane = threadIdx.x & 31;
    return *(const v16bf*)&Wt[(size_t)(((ks * 8 + nt) * 32 + lane) << 4)];
}

// ---- RBF + distance kernel ------------------------------------------------
__global__ void rbf_kernel(const float* __restrict__ R, const int* __restrict__ idx_i,
                           const int* __restrict__ idx_j, const float* __restrict__ centers,
                           const float* __restrict__ widths, bf16* __restrict__ rbf,
                           float* __restrict__ Dij, int P) {
    int p = blockIdx.x * blockDim.x + threadIdx.x;
    if (p >= P) return;
    int i = idx_i[p], j = idx_j[p];
    float dx = R[i * 3 + 0] - R[j * 3 + 0];
    float dy = R[i * 3 + 1] - R[j * 3 + 1];
    float dz = R[i * 3 + 2] - R[j * 3 + 2];
    float d2 = dx * dx + dy * dy + dz * dz;
    float D = sqrtf(fmaxf(d2, 0.0f));
    Dij[p] = D;
    float xc = D * 0.1f;  // D / SR_CUT
    float cf = 0.0f;
    if (xc < 1.0f) {
        float x3 = xc * xc * xc, x4 = x3 * xc, x5 = x4 * xc;
        cf = 1.0f - 6.0f * x5 + 15.0f * x4 - 10.0f * x3;
    }
    float ed = __expf(-D);
    bf16* row = rbf + (size_t)p * K_DIM;
#pragma unroll
    for (int k8 = 0; k8 < K_DIM / 8; ++k8) {
        v8bf vb;
#pragma unroll
        for (int j8 = 0; j8 < 8; ++j8) {
            int k = k8 * 8 + j8;
            float t = ed - centers[k];
            vb[j8] = (bf16)(cf * __expf(-widths[k] * t * t));
        }
        *(v8bf*)&row[k8 * 8] = vb;
    }
}

// ---- init: x = embeddings[Z]; zero accumulators ---------------------------
__global__ void init_kernel(const int* __restrict__ Z, const float* __restrict__ emb,
                            float* __restrict__ x, float* __restrict__ Ea,
                            float* __restrict__ Qa, float* __restrict__ nhAcc, int N) {
    int i = blockIdx.x * blockDim.x + threadIdx.x;
    int total = N * F_DIM;
    if (i < total) {
        int a = i / F_DIM, f = i - a * F_DIM;
        x[i] = emb[(size_t)Z[a] * F_DIM + f];
    }
    if (i < N) { Ea[i] = 0.0f; Qa[i] = 0.0f; }
    if (i == 0) nhAcc[0] = 0.0f;
}

// ---- atom-side GEMM: out = [u.*base +] actA(A) @ W + bias [, ssp] ---------
// 32 rows per wave (two 16-row subtiles sharing each B fragment).
// BASEMODE: 0 none, 1 add base, 2 add uvec[col]*base
template <int ACTA, int ACTOUT, int BASEMODE>
__global__ void __launch_bounds__(128)
gemm_atoms_kernel(const float* __restrict__ A, const float* __restrict__ W,
                  const float* __restrict__ bias, const float* __restrict__ base,
                  const float* __restrict__ uvec, float* __restrict__ out, int N) {
    __shared__ __align__(32) bf16 Wt[4 * 8 * 32 * 16];
    int tid = threadIdx.x;
    fill_wfrag<4>(W, Wt, tid);
    __syncthreads();

    int wave = tid >> 5;
    int rowBase = (blockIdx.x * 4 + wave) * 32;
    if (rowBase >= N) return;
    // second subtile validity (wave-uniform): N is a multiple of 16, so the
    // subtile at rowBase+16 is either fully in bounds or fully out.
    bool two = (rowBase + 16) < N;
    int rb1 = two ? (rowBase + 16) : rowBase;  // safe dummy when !two

    v8f zero = {0.f, 0.f, 0.f, 0.f, 0.f, 0.f, 0.f, 0.f};
    v8f acc[2][8];
#pragma unroll
    for (int nt = 0; nt < 8; ++nt) { acc[0][nt] = zero; acc[1][nt] = zero; }

#pragma unroll
    for (int ks = 0; ks < 4; ++ks) {
        v16bf a0 = load_a_frag_f32<ACTA>(A, rowBase, ks * 32);
        v16bf a1 = load_a_frag_f32<ACTA>(A, rb1, ks * 32);
#pragma unroll
        for (int nt = 0; nt < 8; ++nt) {
            v16bf b = load_b_frag(Wt, ks, nt);
            acc[0][nt] = __builtin_amdgcn_wmma_f32_16x16x32_bf16(false, a0, false, b, (short)0,
                                                                 acc[0][nt], false, false);
            acc[1][nt] = __builtin_amdgcn_wmma_f32_16x16x32_bf16(false, a1, false, b, (short)0,
                                                                 acc[1][nt], false, false);
        }
    }

    int l = tid & 31, half = l >> 4, n16 = l & 15;
#pragma unroll
    for (int rt = 0; rt < 2; ++rt) {
        if (rt == 1 && !two) break;
        int rb = rowBase + rt * 16;
#pragma unroll
        for (int nt = 0; nt < 8; ++nt) {
            int col = nt * 16 + n16;
            float bcol = bias[col];
#pragma unroll
            for (int r = 0; r < 8; ++r) {
                int row = rb + r + half * 8;
                float v = acc[rt][nt][r] + bcol;
                if (ACTOUT) v = ssp(v);
                if (BASEMODE == 1) v += base[(size_t)row * F_DIM + col];
                if (BASEMODE == 2) v += uvec[col] * base[(size_t)row * F_DIM + col];
                out[(size_t)row * F_DIM + col] = v;
            }
        }
    }
}

// ---- fused pair kernel: g = rbf @ k2fW ; msg = g * xjf[idx_j] ;
//      atomic scatter-add into m[idx_i].  32 pair rows per wave. ----------
__global__ void __launch_bounds__(128)
pair_gemm_kernel(const bf16* __restrict__ rbf, const float* __restrict__ k2fW,
                 const int* __restrict__ idx_i, const int* __restrict__ idx_j,
                 const float* __restrict__ xjf, float* __restrict__ m, int P) {
    __shared__ __align__(32) bf16 Wt[2 * 8 * 32 * 16];
    int tid = threadIdx.x;
    fill_wfrag<2>(k2fW, Wt, tid);
    __syncthreads();

    int wave = tid >> 5;
    int rowBase = (blockIdx.x * 4 + wave) * 32;
    if (rowBase >= P) return;

    int l = tid & 31, half = l >> 4, n16 = l & 15;

    // Pull gather/scatter indices early and prefetch their cachelines so the
    // random-access epilogue overlaps with the WMMA work (global_prefetch_b8).
    int ji[16], ii[16];
#pragma unroll
    for (int rt = 0; rt < 2; ++rt) {
#pragma unroll
        for (int r = 0; r < 8; ++r) {
            int row = rowBase + rt * 16 + r + half * 8;
            ji[rt * 8 + r] = idx_j[row];
            ii[rt * 8 + r] = idx_i[row];
            __builtin_prefetch(xjf + (size_t)ji[rt * 8 + r] * F_DIM + n16, 0, 1);
            __builtin_prefetch(m + (size_t)ii[rt * 8 + r] * F_DIM + n16, 1, 1);
        }
    }

    v8f zero = {0.f, 0.f, 0.f, 0.f, 0.f, 0.f, 0.f, 0.f};
    v8f acc[2][8];
#pragma unroll
    for (int nt = 0; nt < 8; ++nt) { acc[0][nt] = zero; acc[1][nt] = zero; }

#pragma unroll
    for (int ks = 0; ks < 2; ++ks) {
        v16bf a0 = load_a_frag_bf16(rbf, rowBase, ks * 32);
        v16bf a1 = load_a_frag_bf16(rbf, rowBase + 16, ks * 32);
#pragma unroll
        for (int nt = 0; nt < 8; ++nt) {
            v16bf b = load_b_frag(Wt, ks, nt);
            acc[0][nt] = __builtin_amdgcn_wmma_f32_16x16x32_bf16(false, a0, false, b, (short)0,
                                                                 acc[0][nt], false, false);
            acc[1][nt] = __builtin_amdgcn_wmma_f32_16x16x32_bf16(false, a1, false, b, (short)0,
                                                                 acc[1][nt], false, false);
        }
    }

#pragma unroll
    for (int rt = 0; rt < 2; ++rt) {
#pragma unroll
        for (int r = 0; r < 8; ++r) {
            const float* xr = xjf + (size_t)ji[rt * 8 + r] * F_DIM;
            float* mr = m + (size_t)ii[rt * 8 + r] * F_DIM;
#pragma unroll
            for (int nt = 0; nt < 8; ++nt) {
                int col = nt * 16 + n16;
                atomicAdd(&mr[col], acc[rt][nt][r] * xr[col]);
            }
        }
    }
}

// ---- output block: out = ssp(o) @ out_W[F,2]; Ea/Qa accumulate; nhloss ---
__global__ void out_block_kernel(const float* __restrict__ o, const float* __restrict__ outW,
                                 float* __restrict__ Ea, float* __restrict__ Qa,
                                 float* __restrict__ cur2, const float* __restrict__ prev2,
                                 float* __restrict__ nhAcc, int useNh, int N) {
    int wave = threadIdx.x >> 5;
    int atom = blockIdx.x * (blockDim.x >> 5) + wave;
    if (atom >= N) return;
    int l = threadIdx.x & 31;
    float s0 = 0.f, s1 = 0.f;
#pragma unroll
    for (int q = 0; q < 4; ++q) {
        int f = l + q * 32;
        float v = ssp(o[(size_t)atom * F_DIM + f]);
        s0 += v * outW[f * 2 + 0];
        s1 += v * outW[f * 2 + 1];
    }
#pragma unroll
    for (int off = 16; off > 0; off >>= 1) {
        s0 += __shfl_down(s0, off, 32);
        s1 += __shfl_down(s1, off, 32);
    }
    if (l == 0) {
        Ea[atom] += s0;
        Qa[atom] += s1;
        float o0 = s0 * s0, o1 = s1 * s1;
        cur2[atom * 2 + 0] = o0;
        cur2[atom * 2 + 1] = o1;
        if (useNh) {
            float p0 = prev2[atom * 2 + 0], p1 = prev2[atom * 2 + 1];
            atomicAdd(nhAcc, o0 / (o0 + p0 + 1e-7f) + o1 / (o1 + p1 + 1e-7f));
        }
    }
}

// ---- finalize: per-element scale/shift, nhloss mean ----------------------
__global__ void finalize_kernel(const int* __restrict__ Z, const float* __restrict__ Es,
                                const float* __restrict__ Esh, const float* __restrict__ Qs,
                                const float* __restrict__ Qsh, const float* __restrict__ Ea,
                                const float* __restrict__ Qa, const float* __restrict__ nhAcc,
                                float* __restrict__ outE, float* __restrict__ outQ,
                                float* __restrict__ outNh, int N) {
    int i = blockIdx.x * blockDim.x + threadIdx.x;
    if (i < N) {
        int z = Z[i];
        outE[i] = Es[z] * Ea[i] + Esh[z];
        outQ[i] = Qs[z] * Qa[i] + Qsh[z];
    }
    if (i == 0) outNh[0] = nhAcc[0] / (2.0f * (float)N);
}

extern "C" void kernel_launch(void* const* d_in, const int* in_sizes, int n_in,
                              void* d_out, int out_size, void* d_ws, size_t ws_size,
                              hipStream_t stream) {
    const int N = in_sizes[0];
    const int P = in_sizes[2];

    const int*   Z       = (const int*)d_in[0];
    const float* R       = (const float*)d_in[1];
    const int*   idx_i   = (const int*)d_in[2];
    const int*   idx_j   = (const int*)d_in[3];
    const float* emb     = (const float*)d_in[4];
    const float* centers = (const float*)d_in[5];
    const float* widths  = (const float*)d_in[6];
    const float* k2fW    = (const float*)d_in[7];
    const float* Wi      = (const float*)d_in[8];
    const float* bi      = (const float*)d_in[9];
    const float* Wj      = (const float*)d_in[10];
    const float* bj      = (const float*)d_in[11];
    const float* riW1    = (const float*)d_in[12];
    const float* rib1    = (const float*)d_in[13];
    const float* riW2    = (const float*)d_in[14];
    const float* rib2    = (const float*)d_in[15];
    const float* dW      = (const float*)d_in[16];
    const float* db      = (const float*)d_in[17];
    const float* u       = (const float*)d_in[18];
    const float* raW1    = (const float*)d_in[19];
    const float* rab1    = (const float*)d_in[20];
    const float* raW2    = (const float*)d_in[21];
    const float* rab2    = (const float*)d_in[22];
    const float* roW1    = (const float*)d_in[23];
    const float* rob1    = (const float*)d_in[24];
    const float* roW2    = (const float*)d_in[25];
    const float* rob2    = (const float*)d_in[26];
    const float* outW    = (const float*)d_in[27];
    const float* Es      = (const float*)d_in[28];
    const float* Esh     = (const float*)d_in[29];
    const float* Qs      = (const float*)d_in[30];
    const float* Qsh     = (const float*)d_in[31];

    float* outE  = (float*)d_out;
    float* outQ  = outE + N;
    float* outD  = outQ + N;      // Dij slice
    float* outNh = outD + P;      // scalar

    // ---- workspace carve-up ----
    char* wptr = (char*)d_ws;
    auto carve = [&](size_t bytes) -> char* {
        char* p = wptr;
        wptr += (bytes + 255) & ~(size_t)255;
        return p;
    };
    bf16*  rbf   = (bf16*)carve((size_t)P * K_DIM * sizeof(bf16));
    float* x     = (float*)carve((size_t)N * F_DIM * sizeof(float));
    float* m     = (float*)carve((size_t)N * F_DIM * sizeof(float));
    float* h     = (float*)carve((size_t)N * F_DIM * sizeof(float));
    float* xjf   = (float*)carve((size_t)N * F_DIM * sizeof(float));
    float* o     = (float*)carve((size_t)N * F_DIM * sizeof(float));
    float* out2a = (float*)carve((size_t)N * 2 * sizeof(float));
    float* out2b = (float*)carve((size_t)N * 2 * sizeof(float));
    float* Ea    = (float*)carve((size_t)N * sizeof(float));
    float* Qa    = (float*)carve((size_t)N * sizeof(float));
    float* nhAcc = (float*)carve(sizeof(float));

    const size_t FF = (size_t)F_DIM * F_DIM;
    dim3 blk128(128), blk256(256);
    int gA = (N + 127) / 128;          // atom GEMM: 4 waves x 32 rows
    int gP = (P + 127) / 128;          // pair GEMM: 4 waves x 32 rows

    rbf_kernel<<<(P + 255) / 256, blk256, 0, stream>>>(R, idx_i, idx_j, centers, widths,
                                                       rbf, outD, P);
    init_kernel<<<(N * F_DIM + 255) / 256, blk256, 0, stream>>>(Z, emb, x, Ea, Qa, nhAcc, N);

    float* cur2 = out2a;
    float* prev2 = out2b;
    for (int b = 0; b < NB; ++b) {
        // xi -> m ; xj -> xjf   (act on input x, act on output)
        gemm_atoms_kernel<1, 1, 0><<<gA, blk128, 0, stream>>>(
            x, Wi + b * FF, bi + (size_t)b * F_DIM, nullptr, nullptr, m, N);
        gemm_atoms_kernel<1, 1, 0><<<gA, blk128, 0, stream>>>(
            x, Wj + b * FF, bj + (size_t)b * F_DIM, nullptr, nullptr, xjf, N);
        // g = rbf @ k2fW ; m[idx_i] += g * xjf[idx_j]
        pair_gemm_kernel<<<gP, blk128, 0, stream>>>(
            rbf, k2fW + (size_t)b * K_DIM * F_DIM, idx_i, idx_j, xjf, m, P);
        // interaction residuals on m
        for (int l = 0; l < NRI; ++l) {
            size_t wOff = ((size_t)b * NRI + l) * FF;
            size_t bOff = ((size_t)b * NRI + l) * F_DIM;
            gemm_atoms_kernel<1, 1, 0><<<gA, blk128, 0, stream>>>(
                m, riW1 + wOff, rib1 + bOff, nullptr, nullptr, h, N);
            gemm_atoms_kernel<0, 0, 1><<<gA, blk128, 0, stream>>>(
                h, riW2 + wOff, rib2 + bOff, m, nullptr, m, N);
        }
        // x = u[b].*x + act(m) @ dW + db
        gemm_atoms_kernel<1, 0, 2><<<gA, blk128, 0, stream>>>(
            m, dW + b * FF, db + (size_t)b * F_DIM, x, u + (size_t)b * F_DIM, x, N);
        // atom residuals on x
        for (int l = 0; l < NRA; ++l) {
            size_t wOff = ((size_t)b * NRA + l) * FF;
            size_t bOff = ((size_t)b * NRA + l) * F_DIM;
            gemm_atoms_kernel<1, 1, 0><<<gA, blk128, 0, stream>>>(
                x, raW1 + wOff, rab1 + bOff, nullptr, nullptr, h, N);
            gemm_atoms_kernel<0, 0, 1><<<gA, blk128, 0, stream>>>(
                h, raW2 + wOff, rab2 + bOff, x, nullptr, x, N);
        }
        // output residual (NRO==1): o = x + act(act(x)@roW1+rob1)@roW2 + rob2
        gemm_atoms_kernel<1, 1, 0><<<gA, blk128, 0, stream>>>(
            x, roW1 + b * FF, rob1 + (size_t)b * F_DIM, nullptr, nullptr, h, N);
        gemm_atoms_kernel<0, 0, 1><<<gA, blk128, 0, stream>>>(
            h, roW2 + b * FF, rob2 + (size_t)b * F_DIM, x, nullptr, o, N);
        // out = ssp(o) @ out_W[b] ; Ea/Qa accumulate ; nhloss term
        out_block_kernel<<<(N + 7) / 8, blk256, 0, stream>>>(
            o, outW + (size_t)b * F_DIM * 2, Ea, Qa, cur2, prev2, nhAcc, (b > 0) ? 1 : 0, N);
        float* t = cur2; cur2 = prev2; prev2 = t;
    }

    finalize_kernel<<<(N + 255) / 256, blk256, 0, stream>>>(Z, Es, Esh, Qs, Qsh, Ea, Qa,
                                                            nhAcc, outE, outQ, outNh, N);
    (void)n_in; (void)out_size; (void)ws_size;
}
